// InferenceVectorQuantizer_19292993093730
// MI455X (gfx1250) — matrix operand order repaired and checked by reading, not compiled, not run on hardware
//
#include <hip/hip_runtime.h>
#include <hip/hip_bf16.h>

typedef __attribute__((ext_vector_type(16))) __bf16 v16bf;
typedef __attribute__((ext_vector_type(8)))  __bf16 v8bf;
typedef __attribute__((ext_vector_type(4)))  __bf16 v4bf;
typedef __attribute__((ext_vector_type(8)))  float  v8f;

#define DDIM   256   // feature dimension (fixed by the reference)
#define CSLICE 4     // codebook slices (occupancy / latency hiding)

// ---------------------------------------------------------------------------
// L2-normalize rows of a [R, 256] f32 matrix, emit bf16 rows.
// One wave per row (8 waves / block). Each lane handles two float4 (8 elems).
// ---------------------------------------------------------------------------
__global__ __launch_bounds__(256) void vq_row_normalize_bf16(
    const float* __restrict__ src, __bf16* __restrict__ dst)
{
  const int lane = threadIdx.x & 31;
  const int wave = threadIdx.x >> 5;
  const size_t row = (size_t)blockIdx.x * 8 + wave;

  const float4* s = (const float4*)(src + row * DDIM);
  float4 a = s[lane];
  float4 b = s[lane + 32];

  float ss = a.x*a.x + a.y*a.y + a.z*a.z + a.w*a.w
           + b.x*b.x + b.y*b.y + b.z*b.z + b.w*b.w;
#pragma unroll
  for (int m = 16; m >= 1; m >>= 1) ss += __shfl_xor(ss, m, 32);

  // reference: x / max(||x||, 1e-8)  ->  x * rsqrt(max(ss, 1e-16))
  const float inv = rsqrtf(fmaxf(ss, 1e-16f));

  v4bf o0 = { (__bf16)(a.x*inv), (__bf16)(a.y*inv),
              (__bf16)(a.z*inv), (__bf16)(a.w*inv) };
  v4bf o1 = { (__bf16)(b.x*inv), (__bf16)(b.y*inv),
              (__bf16)(b.z*inv), (__bf16)(b.w*inv) };
  v4bf* d = (v4bf*)(dst + row * DDIM);
  d[lane]      = o0;
  d[lane + 32] = o1;
}

// ---------------------------------------------------------------------------
// logits = A @ B^T via v_wmma_f32_16x16x32_bf16, fused with running argmax
// over one codebook slice. Per wave: one 16-row M-tile; A stripe in registers
// (8 v16bf). B operands are double-buffered in registers across n-tiles with
// sched_barrier fences so loads for tile nt+1 fly while tile nt's two
// independent WMMA chains (even/odd k) execute.
//
// Layouts used here (A/B: contiguous k order, same in both; C/D as in CDNA5 ISA):
//   A 16x32 bf16 : lane -> row M=lane%16, chunks at kbase, kbase+8,
//                  kbase = kt*32 + (lane/16)*16
//   B 32x16 bf16 : lane -> codebook row n0+lane%16, 32 contiguous bytes at
//                  kt*32 + (lane/16)*16   (B = codebook^T => direct row load)
//   C/D 16x16 f32: acc[j] in lane l = logits[M=j+8*(l/16)][N=l%16]
// ---------------------------------------------------------------------------
__global__ __launch_bounds__(128) void vq_gemm_argmax_slice(
    const __bf16* __restrict__ A,      // [N][256] normalized queries (bf16)
    const __bf16* __restrict__ Bm,     // [C][256] normalized codebook (bf16)
    float*        __restrict__ partV,  // [CSLICE][N] slice max value
    int*          __restrict__ partI,  // [CSLICE][N] slice argmax index
    int N, int C)
{
  const int lane = threadIdx.x & 31;
  const int wave = threadIdx.x >> 5;
  const int half = lane >> 4;           // 0: lanes 0-15, 1: lanes 16-31
  const int lmod = lane & 15;
  const int m0   = blockIdx.x * 64 + wave * 16;   // wave's first row
  const int slc  = blockIdx.y;                    // codebook slice

  const int tilesPerSlice = (C >> 4) / CSLICE;    // 128 (even)
  const int ntBeg = slc * tilesPerSlice;
  const int ntEnd = ntBeg + tilesPerSlice;

  // ---- load A operands (64 VGPRs) ----
  const __bf16* arow = A + (size_t)(m0 + lmod) * DDIM;
  v16bf aop[8];
#pragma unroll
  for (int kt = 0; kt < 8; ++kt) {
    const int kbase = kt * 32 + half * 16;
    v8bf c0 = *(const v8bf*)(arow + kbase);
    v8bf c1 = *(const v8bf*)(arow + kbase + 8);
    aop[kt] = __builtin_shufflevector(c0, c1,
                0,1,2,3,4,5,6,7,8,9,10,11,12,13,14,15);
  }

  float bestv[8];
  int   besti[8];
#pragma unroll
  for (int j = 0; j < 8; ++j) { bestv[j] = -3.4e38f; besti[j] = 0; }

  // B-operand base for a given n-tile (per-lane 32B contiguous chunks)
  auto bbase = [&](int nt) -> const __bf16* {
    return Bm + (size_t)(nt * 16 + lmod) * DDIM + half * 16;
  };
  auto loadB = [&](int nt, v16bf* buf) {
    const __bf16* bptr = bbase(nt);
#pragma unroll
    for (int kt = 0; kt < 8; ++kt)
      buf[kt] = *(const v16bf*)(bptr + kt * 32);
  };
  auto computeTile = [&](const v16bf* buf, int nt) {
    v8f accE = {};
    v8f accO = {};
#pragma unroll
    for (int kt = 0; kt < 8; kt += 2) {
      accE = __builtin_amdgcn_wmma_f32_16x16x32_bf16(
               false, aop[kt],     false, buf[kt],     (short)0, accE,
               false, false);
      accO = __builtin_amdgcn_wmma_f32_16x16x32_bf16(
               false, aop[kt + 1], false, buf[kt + 1], (short)0, accO,
               false, false);
    }
    v8f acc = accE + accO;
    const int col = nt * 16 + lmod;
#pragma unroll
    for (int j = 0; j < 8; ++j) {
      float v = acc[j];
      if (v > bestv[j]) { bestv[j] = v; besti[j] = col; }
    }
  };

  // ---- software-pipelined main loop: register double buffering ----
  v16bf buf0[8], buf1[8];
  loadB(ntBeg, buf0);
  __builtin_amdgcn_sched_barrier(0);

  for (int nt = ntBeg; nt < ntEnd; nt += 2) {
    loadB(nt + 1, buf1);                 // in flight during compute(buf0)
    __builtin_amdgcn_sched_barrier(0);
    computeTile(buf0, nt);
    __builtin_amdgcn_sched_barrier(0);

    const int ntRefill = (nt + 2 < ntEnd) ? (nt + 2) : (ntEnd - 1); // clamp
    loadB(ntRefill, buf0);               // in flight during compute(buf1)
    __builtin_amdgcn_sched_barrier(0);
    computeTile(buf1, nt + 1);
    __builtin_amdgcn_sched_barrier(0);
  }

  // ---- reduce argmax across the 16 column-residue lanes ----
#pragma unroll
  for (int m = 1; m <= 8; m <<= 1) {
#pragma unroll
    for (int j = 0; j < 8; ++j) {
      float ov = __shfl_xor(bestv[j], m, 32);
      int   oi = __shfl_xor(besti[j], m, 32);
      if (ov > bestv[j] || (ov == bestv[j] && oi < besti[j])) {
        bestv[j] = ov; besti[j] = oi;
      }
    }
  }

  if (lmod == 0) {
#pragma unroll
    for (int j = 0; j < 8; ++j) {
      const int grow = m0 + half * 8 + j;
      partV[(size_t)slc * N + grow] = bestv[j];
      partI[(size_t)slc * N + grow] = besti[j];
    }
  }
}

// ---------------------------------------------------------------------------
// Reduce CSLICE partials per row (in increasing-index order -> strict '>'
// preserves first-index tie-break), apply mask, write index as float.
// ---------------------------------------------------------------------------
__global__ __launch_bounds__(256) void vq_finalize(
    const float* __restrict__ partV, const int* __restrict__ partI,
    const int* __restrict__ mask, float* __restrict__ out, int N)
{
  const int row = blockIdx.x * 256 + threadIdx.x;
  if (row >= N) return;
  float bv = partV[row];
  int   bi = partI[row];
#pragma unroll
  for (int s = 1; s < CSLICE; ++s) {
    float v = partV[(size_t)s * N + row];
    int   i = partI[(size_t)s * N + row];
    if (v > bv) { bv = v; bi = i; }
  }
  out[row] = (mask[row] == 0) ? 0.0f : (float)bi;
}

// ---------------------------------------------------------------------------
// Launch: normalize codebook + queries into bf16 workspace, WMMA GEMM+argmax
// over 4 codebook slices, then finalize. Workspace: 16MB + 4MB + 1MB.
// ---------------------------------------------------------------------------
extern "C" void kernel_launch(void* const* d_in, const int* in_sizes, int n_in,
                              void* d_out, int out_size, void* d_ws, size_t ws_size,
                              hipStream_t stream) {
  const float* z_e  = (const float*)d_in[0];   // [8,4096,256] f32
  const int*   mask = (const int*)  d_in[1];   // [8,4096] i32
  const float* cb   = (const float*)d_in[2];   // [8192,256] f32

  const int N = in_sizes[0] / DDIM;            // 32768
  const int C = in_sizes[2] / DDIM;            // 8192

  char* ws = (char*)d_ws;
  __bf16* A_bf  = (__bf16*)ws;                 ws += (size_t)N * DDIM * 2; // 16MB
  __bf16* Cb_bf = (__bf16*)ws;                 ws += (size_t)C * DDIM * 2; //  4MB
  float*  partV = (float*)ws;                  ws += (size_t)CSLICE * N * 4;
  int*    partI = (int*)ws;

  vq_row_normalize_bf16<<<N / 8, 256, 0, stream>>>(z_e, A_bf);
  vq_row_normalize_bf16<<<C / 8, 256, 0, stream>>>(cb,  Cb_bf);

  dim3 grid(N / 64, CSLICE);
  vq_gemm_argmax_slice<<<grid, 128, 0, stream>>>(A_bf, Cb_bf, partV, partI, N, C);

  vq_finalize<<<(N + 255) / 256, 256, 0, stream>>>(partV, partI, mask,
                                                   (float*)d_out, N);
}
